// pair_wise_model_4715874091245
// MI455X (gfx1250) — compile-verified, matrix operand order checked by reading, not run on hardware
//
#include <hip/hip_runtime.h>
#include <hip/hip_bf16.h>

#define DDIM 768
#define NTILES 48                    // 768 / 16 column tiles
#define KSTEPS 24                    // 768 / 32 K steps
#define WAVES_A 8                    // waves per block in kernel A
#define ROWS_PER_WG (WAVES_A * 16)   // 128 rows per workgroup
#define WELEM (DDIM * DDIM)          // elements in one Wd matrix
#define STRIP_ELEM (16 * DDIM)       // 12288 bf16 elements per 16-row strip

typedef __attribute__((ext_vector_type(16))) __bf16 v16bf;
typedef __attribute__((ext_vector_type(8)))  float  v8f;
typedef __attribute__((ext_vector_type(4)))  unsigned int u32x4;
typedef __attribute__((ext_vector_type(8)))  int i32x8;
typedef __attribute__((ext_vector_type(4)))  int i32x4;

union FragU {
    uint4  u[2];
    v16bf  v;
};

// round-to-nearest-even f32 -> bf16, packed pair into one dword
__device__ __forceinline__ unsigned int pack_bf16(float lo, float hi) {
    unsigned int ul = __float_as_uint(lo);
    unsigned int uh = __float_as_uint(hi);
    ul = (ul + 0x7FFFu + ((ul >> 16) & 1u)) >> 16;
    uh = (uh + 0x7FFFu + ((uh >> 16) & 1u)) >> 16;
    return ul | (uh << 16);
}

// 16x32 bf16 A/B fragment from LDS (ISA 7.12.2): lane L holds row (L&15);
// elements 0..7 = K khalf*8+0..7, elements 8..15 = K 16+khalf*8+0..7.
__device__ __forceinline__ v16bf load_frag_lds(const unsigned short* p) {
    FragU f;
    f.u[0] = *(const uint4*)(p);        // first 8 bf16 (16B)
    f.u[1] = *(const uint4*)(p + 16);   // +16 elements = +32B
    return f.v;
}

// Issue a TDM 1-D copy of one 16x768 bf16 strip (24 KB, contiguous) into LDS.
// D# group0: count=1 | lds_addr | global_addr | type=2.
// D# group1: data_size=2B, tensor_dim0=tile_dim0=stride0=12288, tensor_dim1=1.
__device__ __forceinline__ void tdm_load_strip(const unsigned short* gsrc,
                                               const unsigned short* ldst) {
    unsigned long long ga = (unsigned long long)gsrc;
    u32x4 g0;
    g0.x = 1u;                                            // count=1 (valid)
    g0.y = (unsigned int)(unsigned long long)ldst;        // lds_addr (bytes)
    g0.z = (unsigned int)ga;                              // global_addr[31:0]
    g0.w = (unsigned int)((ga >> 32) & 0x01FFFFFFull)     // global_addr[56:32]
         | 0x80000000u;                                   // type=2 ("image")
    i32x8 g1 = { 0x00010000,   // data_size=1 (2 bytes/elem), mask=0
                 0x30000000,   // tensor_dim0[15:0]=12288 << 16
                 0x00010000,   // tensor_dim1[15:0]=1 << 16
                 0x30000000,   // tile_dim0=12288 << 16
                 0,            // tile_dim1=0, tile_dim2=0
                 STRIP_ELEM,   // tensor_dim0_stride = 12288
                 0, 0 };
    i32x4 gz = {0, 0, 0, 0};
#if defined(__clang_major__) && (__clang_major__ >= 23)
    i32x8 gz8 = {0, 0, 0, 0, 0, 0, 0, 0};
    __builtin_amdgcn_tensor_load_to_lds(g0, g1, gz, gz, gz8, 0);
#else
    __builtin_amdgcn_tensor_load_to_lds(g0, g1, gz, gz, 0);
#endif
}

// Prep: fp32 -> bf16 conversion (Wd matrices into workspace)
__global__ __launch_bounds__(256) void convert_bf16(const float* __restrict__ src,
                                                    unsigned short* __restrict__ dst,
                                                    int n4) {
    int i = blockIdx.x * blockDim.x + threadIdx.x;
    if (i < n4) {
        float4 x = ((const float4*)src)[i];
        uint2 p;
        p.x = pack_bf16(x.x, x.y);
        p.y = pack_bf16(x.z, x.w);
        ((uint2*)dst)[i] = p;
    }
}

__global__ __launch_bounds__(256) void fused_dense_tanh_proj(
    const float* __restrict__ Xb, const float* __restrict__ Xa,
    const unsigned short* __restrict__ wsW,   // [2][768*768] bf16 (bert, albert)
    const float* __restrict__ bd_b, const float* __restrict__ bd_a,
    const float* __restrict__ Wp_b, const float* __restrict__ Wp_a,
    const float* __restrict__ bp_b, const float* __restrict__ bp_a,
    const float* __restrict__ b1_b, const float* __restrict__ b2_b,
    const float* __restrict__ b1_a, const float* __restrict__ b2_a,
    float* __restrict__ out)
{
    extern __shared__ char smem_raw[];
    // layout: [WAVES_A][16][DDIM] bf16 X tiles, then 2 ping-pong Wd strips.
    // Single base pointers + integer offsets only, so addrspace(3) inference
    // holds and fragment reads lower to ds_load_b128 (not flat_load).
    unsigned short* ldsX = (unsigned short*)smem_raw;
    unsigned short* ldsWbase = ldsX + (size_t)WAVES_A * STRIP_ELEM;

    const int tid  = threadIdx.x;
    const int wave = tid >> 5;
    const int lane = tid & 31;
    const int ln   = lane & 15;     // N (B/C) or M (A) index within tile
    const int kh   = lane >> 4;     // K-half selector
    const int rowBase = blockIdx.x * ROWS_PER_WG;

    const float* Xm[2]  = {Xb, Xa};
    const float* bdm[2] = {bd_b, bd_a};
    const float* Wpm[2] = {Wp_b, Wp_a};

    float a0[8], a1[8];
#pragma unroll
    for (int v = 0; v < 8; ++v) { a0[v] = 0.0f; a1[v] = 0.0f; }

    // prologue: DMA strip 0 (bert, n-tile 0) into buffer 0
    if (wave == 0) {
        tdm_load_strip(wsW, ldsWbase);
        __builtin_amdgcn_s_wait_tensorcnt(0);
    }

#pragma unroll 1
    for (int s = 0; s < 2 * NTILES; ++s) {
        const int mod = s / NTILES;
        const int nt  = s - mod * NTILES;
        const int n0  = nt * 16;

        // ---- at model boundary: stage this wave's 16xD X tile (fp32->bf16).
        // ldsX region is wave-private; per-wave DS ordering makes it safe.
        if (nt == 0) {
            const float* gx = Xm[mod] + (size_t)(rowBase + wave * 16) * DDIM;
            unsigned short* dst = ldsX + wave * STRIP_ELEM;
#pragma unroll 4
            for (int i = 0; i < STRIP_ELEM / (32 * 4); ++i) {   // 96 iters
                int e = (i * 32 + lane) * 4;
                float4 x = *(const float4*)(gx + e);
                uint2 p;
                p.x = pack_bf16(x.x, x.y);
                p.y = pack_bf16(x.z, x.w);
                *(uint2*)(dst + e) = p;
            }
        }

        // publish strip s to all waves; also proves all readers of buffer
        // (s+1)&1 from iteration s-1 are done before we overwrite it
        __syncthreads();

        // ---- issue async DMA for strip s+1 into the other buffer (overlaps WMMA)
        if (wave == 0 && (s + 1) < 2 * NTILES) {
            const int sn   = s + 1;
            const int smod = sn / NTILES;
            const int snt  = sn - smod * NTILES;
            tdm_load_strip(wsW + (size_t)smod * WELEM + (size_t)snt * STRIP_ELEM,
                           ldsWbase + (size_t)(sn & 1) * STRIP_ELEM);
        }

        // ---- 16x16 f32 accumulator over K = 0..767 via bf16 WMMA
        v8f c = {0.f, 0.f, 0.f, 0.f, 0.f, 0.f, 0.f, 0.f};
        const unsigned short* xbase = ldsX + wave * STRIP_ELEM + ln * DDIM + kh * 8;
        const unsigned short* wbase = ldsWbase + (unsigned)(s & 1) * STRIP_ELEM
                                      + ln * DDIM + kh * 8;
#pragma unroll
        for (int ks = 0; ks < KSTEPS; ++ks) {
            v16bf a = load_frag_lds(xbase + ks * 32);
            v16bf b = load_frag_lds(wbase + ks * 32);
            c = __builtin_amdgcn_wmma_f32_16x16x32_bf16(
                    false, a, false, b, (short)0, c, false, false);
        }

        // ---- fused tanh + 2-column projection; C layout: VGPR v, lane L:
        //      M = v + 8*(L>>4), N = L&15
        const float bdv = bdm[mod][n0 + ln];
        const float wp0 = Wpm[mod][n0 + ln];
        const float wp1 = Wpm[mod][DDIM + n0 + ln];
#pragma unroll
        for (int v = 0; v < 8; ++v) {
            float t = tanhf(c[v] + bdv);
            a0[v] += t * wp0;
            a1[v] += t * wp1;
        }

        // strip s+1 must be fully in LDS before next iteration's barrier
        if (wave == 0) __builtin_amdgcn_s_wait_tensorcnt(0);
    }

    // ---- reduce across the 16 N-lanes of each half-wave
#pragma unroll
    for (int v = 0; v < 8; ++v) {
#pragma unroll
        for (int msk = 8; msk; msk >>= 1) {
            a0[v] += __shfl_xor(a0[v], msk, 16);
            a1[v] += __shfl_xor(a1[v], msk, 16);
        }
    }

    if (ln == 0) {
        const float cb0 = bp_b[0] + bp_a[0] + b1_b[0] + b2_b[0] + b1_a[0] + b2_a[0];
        const float cb1 = bp_b[1] + bp_a[1] + b1_b[1] + b2_b[1] + b1_a[1] + b2_a[1];
#pragma unroll
        for (int v = 0; v < 8; ++v) {
            int row = rowBase + wave * 16 + kh * 8 + v;
            out[(size_t)row * 2 + 0] = a0[v] + cb0;
            out[(size_t)row * 2 + 1] = a1[v] + cb1;
        }
    }
}

// Gather terms: out[m] += db_j@W1_b.T + db_k@W2_b.T + da_j@(W1_a+W2_a).T
// (reference uses da_j for BOTH albert terms — replicated exactly)
__global__ __launch_bounds__(256) void gather_proj(
    const float* __restrict__ data_b, const float* __restrict__ data_a,
    const int* __restrict__ pairs,
    const float* __restrict__ W1_b, const float* __restrict__ W2_b,
    const float* __restrict__ W1_a, const float* __restrict__ W2_a,
    float* __restrict__ out, int Btot)
{
    const int wave = threadIdx.x >> 5;
    const int lane = threadIdx.x & 31;
    const int m = blockIdx.x * 8 + wave;
    if (m >= Btot) return;

    const int idx = pairs[3 * m + 0];
    const int j   = pairs[3 * m + 1];
    const int kk  = pairs[3 * m + 2];
    const float* dbj = data_b + ((size_t)idx * 64 + j)  * DDIM;
    const float* dbk = data_b + ((size_t)idx * 64 + kk) * DDIM;
    const float* daj = data_a + ((size_t)idx * 64 + j)  * DDIM;

    float s0 = 0.f, s1 = 0.f;
#pragma unroll 4
    for (int t = 0; t < DDIM / 32; ++t) {
        int k = lane + 32 * t;
        float vbj = dbj[k], vbk = dbk[k], vaj = daj[k];
        float wa0 = W1_a[k] + W2_a[k];
        float wa1 = W1_a[DDIM + k] + W2_a[DDIM + k];
        s0 += vbj * W1_b[k]        + vbk * W2_b[k]        + vaj * wa0;
        s1 += vbj * W1_b[DDIM + k] + vbk * W2_b[DDIM + k] + vaj * wa1;
    }
#pragma unroll
    for (int msk = 16; msk; msk >>= 1) {
        s0 += __shfl_xor(s0, msk, 32);
        s1 += __shfl_xor(s1, msk, 32);
    }
    if (lane == 0) {
        out[(size_t)m * 2 + 0] += s0;
        out[(size_t)m * 2 + 1] += s1;
    }
}

extern "C" void kernel_launch(void* const* d_in, const int* in_sizes, int n_in,
                              void* d_out, int out_size, void* d_ws, size_t ws_size,
                              hipStream_t stream) {
    const float* Xb     = (const float*)d_in[0];   // pair_embeddings_bert   (B, D)
    const float* data_b = (const float*)d_in[1];   // data_after_level_two_bert (512,64,D)
    const int*   pairs  = (const int*)  d_in[2];   // (B, 3)
    const float* Xa     = (const float*)d_in[3];   // pair_embeddings_albert (B, D)
    const float* data_a = (const float*)d_in[4];   // data_after_level_two_albert
    const float* Wd_b   = (const float*)d_in[5];
    const float* bd_b   = (const float*)d_in[6];
    const float* Wd_a   = (const float*)d_in[7];
    const float* bd_a   = (const float*)d_in[8];
    const float* Wp_b   = (const float*)d_in[9];
    const float* bp_b   = (const float*)d_in[10];
    const float* Wp_a   = (const float*)d_in[11];
    const float* bp_a   = (const float*)d_in[12];
    const float* W1_b   = (const float*)d_in[13];
    const float* b1_b   = (const float*)d_in[14];
    const float* W2_b   = (const float*)d_in[15];
    const float* b2_b   = (const float*)d_in[16];
    const float* W1_a   = (const float*)d_in[17];
    const float* b1_a   = (const float*)d_in[18];
    const float* W2_a   = (const float*)d_in[19];
    const float* b2_a   = (const float*)d_in[20];
    float* out = (float*)d_out;

    const int Btot = in_sizes[0] / DDIM;           // 65536

    // Prep: bf16 copies of Wd_b / Wd_a into workspace (2 x 1.125 MB)
    unsigned short* wsW = (unsigned short*)d_ws;
    {
        const int n4 = WELEM / 4;
        dim3 gridP((n4 + 255) / 256);
        convert_bf16<<<gridP, 256, 0, stream>>>(Wd_b, wsW, n4);
        convert_bf16<<<gridP, 256, 0, stream>>>(Wd_a, wsW + WELEM, n4);
    }

    // Kernel A: fused dense + tanh + projection, TDM double-buffered Wd strips
    const size_t ldsBytes = ((size_t)WAVES_A * STRIP_ELEM + 2 * STRIP_ELEM) * 2; // 240 KB
    dim3 gridA(Btot / ROWS_PER_WG);
    fused_dense_tanh_proj<<<gridA, 256, ldsBytes, stream>>>(
        Xb, Xa, wsW, bd_b, bd_a, Wp_b, Wp_a,
        bp_b, bp_a, b1_b, b2_b, b1_a, b2_a, out);

    // Kernel B: gather-row projections accumulated on top (in-order stream)
    dim3 gridB((Btot + 7) / 8);
    gather_proj<<<gridB, 256, 0, stream>>>(
        data_b, data_a, pairs, W1_b, W2_b, W1_a, W2_a, out, Btot);
}
